// GPT_84121229459709
// MI455X (gfx1250) — matrix operand order repaired and checked by reading, not compiled, not run on hardware
//
#include <hip/hip_runtime.h>
#include <hip/hip_bf16.h>
#include <math.h>

// ---- problem constants (GPT-2 small, B=4, T=512) ----
#define NL  12
#define NC  768
#define NH  12
#define NFF 3072
#define NV  50257
#define NT  512
#define NB  4
#define ND  64
#define BT  (NB * NT)   // 2048
#define C3  (3 * NC)    // 2304

typedef __bf16 bf16;
typedef __attribute__((ext_vector_type(16))) __bf16 v16bf;
typedef __attribute__((ext_vector_type(8)))  __bf16 v8bf;
typedef __attribute__((ext_vector_type(8)))  float  v8f;

// ---------------------------------------------------------------------------
// Generic batched WMMA GEMM:  C[z] = act( scale * A[z] @ Bt[z]^T + bias ) + res
//   A  : [M,K]  bf16 row-major (lda)
//   Bt : [N,K]  bf16 row-major (ldb)  == B column-major, WMMA-friendly
//   z  : zb = z / H, zh = z % H, two-component batch strides
//   out: f32 (Cf) or bf16 (Cb), optional f32 residual added before store
// Each wave computes a 32x64 tile: 2 A frags x 4 B frags -> 8 WMMAs / K-step.
// K loop is software-pipelined with two fragment sets (requires K/32 even,
// true for all GEMMs here: K in {64, 512, 768, 3072}).
// BOUND=false: N multiple of 64, no column checks in the hot loop.
// BOUND=true : clamp B pointers to col N-1 (loads legal, stores guarded).
// ---------------------------------------------------------------------------
struct GemmArgs {
  const bf16* A;  const bf16* Bt; const float* bias;
  float* Cf;      bf16* Cb;       const float* res;
  int M, N, K, lda, ldb, ldc;
  int H;
  long aZ0, aZ1, bZ0, bZ1, cZ0, cZ1;
  float scale;
  int act;   // 0 = none, 1 = tanh-GELU
};

__device__ __forceinline__ float gelu_tanh(float x) {
  float x3 = x * x * x;
  return 0.5f * x * (1.0f + tanhf(0.79788456080286535588f * (x + 0.044715f * x3)));
}

struct Frag {
  v16bf a0, a1;
  v16bf b[4];
};

__device__ __forceinline__ void load_frag(Frag& f,
                                          const bf16* arow0, const bf16* arow1,
                                          const bf16* const* brow, int k0) {
  // A fragments: 16x32 bf16 each; lane r holds K = kg*8..+7 and 16+kg*8..+7
  *(v8bf*)&f.a0       = *(const v8bf*)(arow0 + k0);
  *((v8bf*)&f.a0 + 1) = *(const v8bf*)(arow0 + k0 + 16);
  *(v8bf*)&f.a1       = *(const v8bf*)(arow1 + k0);
  *((v8bf*)&f.a1 + 1) = *(const v8bf*)(arow1 + k0 + 16);
  // B fragments: 32x16 bf16; lane col holds K = kg*8..+7 and 16+kg*8..+7
#pragma unroll
  for (int s = 0; s < 4; ++s) {
    *(v8bf*)&f.b[s]       = *(const v8bf*)(brow[s] + k0);
    *((v8bf*)&f.b[s] + 1) = *(const v8bf*)(brow[s] + k0 + 16);
  }
  __builtin_prefetch(arow0 + k0 + 256, 0, 1);   // global_prefetch_b8
}

__device__ __forceinline__ void do_wmma(const Frag& f, v8f acc[2][4]) {
#pragma unroll
  for (int s = 0; s < 4; ++s) {
    acc[0][s] = __builtin_amdgcn_wmma_f32_16x16x32_bf16(
        false, f.a0, false, f.b[s], (short)0, acc[0][s], false, false);
    acc[1][s] = __builtin_amdgcn_wmma_f32_16x16x32_bf16(
        false, f.a1, false, f.b[s], (short)0, acc[1][s], false, false);
  }
}

template <bool BOUND>
__global__ __launch_bounds__(256) void gemm_bf16_wmma(GemmArgs g) {
  const int tid  = threadIdx.x;
  const int lane = tid & 31;
  const int wave = tid >> 5;

  const int z  = blockIdx.z;
  const int zb = z / g.H;
  const int zh = z % g.H;

  const bf16* A  = g.A  + (size_t)zb * g.aZ0 + (size_t)zh * g.aZ1;
  const bf16* Bt = g.Bt + (size_t)zb * g.bZ0 + (size_t)zh * g.bZ1;
  const size_t cOff = (size_t)zb * g.cZ0 + (size_t)zh * g.cZ1;

  const int m0 = blockIdx.y * 256 + wave * 32;   // M multiple of 256
  const int n0 = blockIdx.x * 64;

  const int r  = lane & 15;   // A row in tile / B column in subtile
  const int kg = lane >> 4;   // K half selector

  v8f acc[2][4] = {};

  // Loop-invariant base pointers (only +k0 varies in the hot loop).
  const bf16* arow0 = A + (size_t)(m0 + r) * g.lda + kg * 8;        // rows m0..m0+15
  const bf16* arow1 = arow0 + (size_t)16 * g.lda;                   // rows m0+16..m0+31
  const bf16* brow[4];
#pragma unroll
  for (int s = 0; s < 4; ++s) {
    int col = n0 + s * 16 + r;
    if (BOUND) col = (col < g.N) ? col : (g.N - 1);   // clamp, never branch
    brow[s] = Bt + (size_t)col * g.ldb + kg * 8;
  }

  // Software-pipelined K loop (double-buffered; K/32 is always even here).
  Frag f0, f1;
  load_frag(f0, arow0, arow1, brow, 0);
  int k0 = 0;
  for (; k0 + 64 < g.K; k0 += 64) {
    load_frag(f1, arow0, arow1, brow, k0 + 32);
    do_wmma(f0, acc);
    load_frag(f0, arow0, arow1, brow, k0 + 64);
    do_wmma(f1, acc);
  }
  load_frag(f1, arow0, arow1, brow, k0 + 32);
  do_wmma(f0, acc);
  do_wmma(f1, acc);

  // C/D layout: VGPR i -> row (tile base) + i + 8*kg, col = n0 + s*16 + r
#pragma unroll
  for (int t = 0; t < 2; ++t) {
#pragma unroll
    for (int s = 0; s < 4; ++s) {
      const int col = n0 + s * 16 + r;
      if (BOUND && col >= g.N) continue;
      const float bv = g.bias ? g.bias[col] : 0.0f;
#pragma unroll
      for (int i = 0; i < 8; ++i) {
        const int row = m0 + t * 16 + kg * 8 + i;
        float v = acc[t][s][i] * g.scale + bv;
        if (g.act == 1) v = gelu_tanh(v);
        const size_t cidx = cOff + (size_t)row * g.ldc + col;
        if (g.res) v += g.res[cidx];
        if (g.Cf) g.Cf[cidx] = v;
        else      g.Cb[cidx] = (bf16)v;
      }
    }
  }
}

// ---------------------------------------------------------------------------
// Weight convert + transpose:  Wt[l][n][k] = bf16( W[l][k][n] )
// ---------------------------------------------------------------------------
__global__ __launch_bounds__(256) void convT_kernel(const float* __restrict__ W,
                                                    bf16* __restrict__ Wt,
                                                    int K, int N, int Lc) {
  const long i   = (long)blockIdx.x * 256 + threadIdx.x;
  const long per = (long)K * N;
  if (i >= per * Lc) return;
  const int  l = (int)(i / per);
  const long rm = i % per;
  const int  k = (int)(rm / N);
  const int  n = (int)(rm % N);
  Wt[(size_t)l * per + (size_t)n * K + k] = (bf16)W[i];
}

// ---------------------------------------------------------------------------
// Embedding:  x[row,c] = wte[idx[row],c] + wpe[row % T, c]
// ---------------------------------------------------------------------------
__global__ __launch_bounds__(256) void embed_kernel(const int* __restrict__ idx,
                                                    const float* __restrict__ wte,
                                                    const float* __restrict__ wpe,
                                                    float* __restrict__ x) {
  const int i = blockIdx.x * 256 + threadIdx.x;
  if (i >= BT * NC) return;
  const int row = i / NC, c = i % NC;
  const int t = row % NT;
  x[i] = wte[(size_t)idx[row] * NC + c] + wpe[(size_t)t * NC + c];
}

// ---------------------------------------------------------------------------
// LayerNorm (f32 in -> bf16 out), one block per row of length NC
// ---------------------------------------------------------------------------
__global__ __launch_bounds__(256) void layernorm_kernel(const float* __restrict__ x,
                                                        const float* __restrict__ w,
                                                        const float* __restrict__ b,
                                                        bf16* __restrict__ out) {
  __shared__ float red[256];
  const int row = blockIdx.x;
  const int tid = threadIdx.x;
  const float* xr = x + (size_t)row * NC;

  float s = 0.f;
  for (int c = tid; c < NC; c += 256) s += xr[c];
  red[tid] = s; __syncthreads();
  for (int st = 128; st > 0; st >>= 1) { if (tid < st) red[tid] += red[tid + st]; __syncthreads(); }
  const float mean = red[0] * (1.0f / NC);
  __syncthreads();

  float vs = 0.f;
  for (int c = tid; c < NC; c += 256) { float d = xr[c] - mean; vs += d * d; }
  red[tid] = vs; __syncthreads();
  for (int st = 128; st > 0; st >>= 1) { if (tid < st) red[tid] += red[tid + st]; __syncthreads(); }
  const float inv = rsqrtf(red[0] * (1.0f / NC) + 1e-5f);

  for (int c = tid; c < NC; c += 256)
    out[(size_t)row * NC + c] = (bf16)((xr[c] - mean) * inv * w[c] + b[c]);
}

// ---------------------------------------------------------------------------
// Transpose V:  vt[(b*H+h)*D + d][t] = qkv[b*T+t][2C + h*D + d]   (bf16)
// ---------------------------------------------------------------------------
__global__ __launch_bounds__(256) void vtrans_kernel(const bf16* __restrict__ qkv,
                                                     bf16* __restrict__ vt) {
  const int i = blockIdx.x * 256 + threadIdx.x;
  if (i >= NB * NH * ND * NT) return;
  const int t  = i % NT;
  const int rm = i / NT;
  const int d  = rm % ND;
  const int bh = rm / ND;
  const int b  = bh / NH, h = bh % NH;
  vt[i] = qkv[(size_t)(b * NT + t) * C3 + 2 * NC + h * ND + d];
}

// ---------------------------------------------------------------------------
// Causal softmax over fp32 scores -> bf16 P.  grid = (T, B*H), row q = blockIdx.x
// ---------------------------------------------------------------------------
__global__ __launch_bounds__(256) void softmax_kernel(const float* __restrict__ sc,
                                                      bf16* __restrict__ P) {
  __shared__ float red[256];
  const int q = blockIdx.x;
  const int z = blockIdx.y;
  const int tid = threadIdx.x;
  const float* sr = sc + ((size_t)z * NT + q) * NT;
  bf16* pr = P + ((size_t)z * NT + q) * NT;

  float m = -1e30f;
  for (int k = tid; k <= q; k += 256) m = fmaxf(m, sr[k]);
  red[tid] = m; __syncthreads();
  for (int st = 128; st > 0; st >>= 1) { if (tid < st) red[tid] = fmaxf(red[tid], red[tid + st]); __syncthreads(); }
  const float mx = red[0]; __syncthreads();

  float ssum = 0.f;
  for (int k = tid; k <= q; k += 256) ssum += __expf(sr[k] - mx);
  red[tid] = ssum; __syncthreads();
  for (int st = 128; st > 0; st >>= 1) { if (tid < st) red[tid] += red[tid + st]; __syncthreads(); }
  const float inv = 1.0f / red[0];

  for (int k = tid; k < NT; k += 256)
    pr[k] = (k <= q) ? (bf16)(__expf(sr[k] - mx) * inv) : (bf16)0.0f;
}

// ---------------------------------------------------------------------------
// Cross entropy:  per-row log-sum-exp + NLL accumulation
// ---------------------------------------------------------------------------
__global__ __launch_bounds__(256) void ce_row_kernel(const float* __restrict__ logits,
                                                     const int* __restrict__ tgt,
                                                     float* __restrict__ accum) {
  __shared__ float red[256];
  const int row = blockIdx.x;
  const int tid = threadIdx.x;
  const float* lr = logits + (size_t)row * NV;

  float m = -1e30f;
  for (int c = tid; c < NV; c += 256) m = fmaxf(m, lr[c]);
  red[tid] = m; __syncthreads();
  for (int st = 128; st > 0; st >>= 1) { if (tid < st) red[tid] = fmaxf(red[tid], red[tid + st]); __syncthreads(); }
  const float mx = red[0]; __syncthreads();

  float s = 0.f;
  for (int c = tid; c < NV; c += 256) s += __expf(lr[c] - mx);
  red[tid] = s; __syncthreads();
  for (int st = 128; st > 0; st >>= 1) { if (tid < st) red[tid] += red[tid + st]; __syncthreads(); }

  if (tid == 0) {
    const int t = tgt[row];
    if (t != -100) {
      const float lse = mx + __logf(red[0]);
      atomicAdd(&accum[0], lse - lr[t]);
      atomicAdd(&accum[1], 1.0f);
    }
  }
}

__global__ void zero_accum_kernel(float* a) { a[threadIdx.x] = 0.0f; }

__global__ void finalize_loss_kernel(const float* __restrict__ accum,
                                     float* __restrict__ out) {
  out[0] = accum[0] / fmaxf(accum[1], 1.0f);
}

// ---------------------------------------------------------------------------
// Host: orchestrate the full forward pass on `stream`
// ---------------------------------------------------------------------------
extern "C" void kernel_launch(void* const* d_in, const int* in_sizes, int n_in,
                              void* d_out, int out_size, void* d_ws, size_t ws_size,
                              hipStream_t stream) {
  (void)in_sizes; (void)n_in; (void)out_size; (void)ws_size;

  const int*   idx     = (const int*)  d_in[0];
  const int*   targets = (const int*)  d_in[1];
  const float* wte     = (const float*)d_in[2];
  const float* wpe     = (const float*)d_in[3];
  const float* ln1_w   = (const float*)d_in[4];
  const float* ln1_b   = (const float*)d_in[5];
  const float* attn_w  = (const float*)d_in[6];
  const float* attn_b  = (const float*)d_in[7];
  const float* attnp_w = (const float*)d_in[8];
  const float* attnp_b = (const float*)d_in[9];
  const float* ln2_w   = (const float*)d_in[10];
  const float* ln2_b   = (const float*)d_in[11];
  const float* fc_w    = (const float*)d_in[12];
  const float* fc_b    = (const float*)d_in[13];
  const float* fcp_w   = (const float*)d_in[14];
  const float* fcp_b   = (const float*)d_in[15];
  const float* lnf_w   = (const float*)d_in[16];
  const float* lnf_b   = (const float*)d_in[17];
  const float* lm_w    = (const float*)d_in[18];
  float* out = (float*)d_out;   // [BT*NV logits] ++ [1 loss]

  // ---- workspace carving ----
  char* p = (char*)d_ws;
  auto carve = [&](size_t bytes) -> char* {
    char* r = p; p += (bytes + 255) & ~(size_t)255; return r;
  };
  bf16*  awT  = (bf16*) carve((size_t)NL * C3  * NC  * 2);
  bf16*  apT  = (bf16*) carve((size_t)NL * NC  * NC  * 2);
  bf16*  fcT  = (bf16*) carve((size_t)NL * NFF * NC  * 2);
  bf16*  fpT  = (bf16*) carve((size_t)NL * NC  * NFF * 2);
  bf16*  lmT  = (bf16*) carve((size_t)NV * NC  * 2);
  float* x    = (float*)carve((size_t)BT * NC * 4);
  bf16*  h    = (bf16*) carve((size_t)BT * NC * 2);
  bf16*  qkv  = (bf16*) carve((size_t)BT * C3 * 2);
  bf16*  vt   = (bf16*) carve((size_t)NB * NH * ND * NT * 2);
  float* sc   = (float*)carve((size_t)NB * NH * NT * NT * 4);
  bf16*  P    = (bf16*) carve((size_t)NB * NH * NT * NT * 2);
  bf16*  y    = (bf16*) carve((size_t)BT * NC * 2);
  bf16*  ff   = (bf16*) carve((size_t)BT * NFF * 2);
  float* acc  = (float*)carve(64);

  auto blocks1d = [](long total) { return dim3((unsigned)((total + 255) / 256)); };

  auto gemm = [&](const bf16* A, long aZ0, long aZ1, int lda,
                  const bf16* Bt, long bZ0, long bZ1, int ldb,
                  const float* bias, float* Cf, bf16* Cb, const float* res,
                  long cZ0, long cZ1, int ldc,
                  int M, int N, int K, int H, int Z, float scale, int act,
                  bool bounded) {
    GemmArgs g;
    g.A = A; g.Bt = Bt; g.bias = bias; g.Cf = Cf; g.Cb = Cb; g.res = res;
    g.M = M; g.N = N; g.K = K; g.lda = lda; g.ldb = ldb; g.ldc = ldc;
    g.H = H; g.aZ0 = aZ0; g.aZ1 = aZ1; g.bZ0 = bZ0; g.bZ1 = bZ1;
    g.cZ0 = cZ0; g.cZ1 = cZ1; g.scale = scale; g.act = act;
    dim3 grid((N + 63) / 64, M / 256, Z);
    if (bounded) gemm_bf16_wmma<true ><<<grid, 256, 0, stream>>>(g);
    else         gemm_bf16_wmma<false><<<grid, 256, 0, stream>>>(g);
  };

  // ---- one-time weight conversion (bf16 + transpose to N-major) ----
  convT_kernel<<<blocks1d((long)NL * NC * C3),  256, 0, stream>>>(attn_w,  awT, NC,  C3,  NL);
  convT_kernel<<<blocks1d((long)NL * NC * NC),  256, 0, stream>>>(attnp_w, apT, NC,  NC,  NL);
  convT_kernel<<<blocks1d((long)NL * NC * NFF), 256, 0, stream>>>(fc_w,    fcT, NC,  NFF, NL);
  convT_kernel<<<blocks1d((long)NL * NFF * NC), 256, 0, stream>>>(fcp_w,   fpT, NFF, NC,  NL);
  convT_kernel<<<blocks1d((long)NC * NV),       256, 0, stream>>>(lm_w,    lmT, NC,  NV,  1);

  // ---- embeddings ----
  embed_kernel<<<blocks1d((long)BT * NC), 256, 0, stream>>>(idx, wte, wpe, x);

  // ---- transformer blocks ----
  for (int l = 0; l < NL; ++l) {
    layernorm_kernel<<<BT, 256, 0, stream>>>(x, ln1_w + l * NC, ln1_b + l * NC, h);

    // qkv = h @ attn_w + attn_b   -> bf16 [BT, 3C]
    gemm(h, 0, 0, NC,
         awT + (size_t)l * C3 * NC, 0, 0, NC,
         attn_b + l * C3, nullptr, qkv, nullptr,
         0, 0, C3, BT, C3, NC, 1, 1, 1.0f, 0, false);

    vtrans_kernel<<<blocks1d((long)NB * NH * ND * NT), 256, 0, stream>>>(qkv, vt);

    // scores[z] = (Q @ K^T) / 8   (z = b*H + h), K row-major == N-major B
    gemm(qkv,      (long)NT * C3, ND, C3,
         qkv + NC, (long)NT * C3, ND, C3,
         nullptr, sc, nullptr, nullptr,
         (long)NH * NT * NT, (long)NT * NT, NT,
         NT, NT, ND, NH, NB * NH, 0.125f, 0, false);

    softmax_kernel<<<dim3(NT, NB * NH), 256, 0, stream>>>(sc, P);

    // y[z] = P @ V   scattered back into [BT, C] at column h*64
    gemm(P,  (long)NH * NT * NT, (long)NT * NT, NT,
         vt, (long)NH * ND * NT, (long)ND * NT, NT,
         nullptr, nullptr, y, nullptr,
         (long)NT * NC, ND, NC,
         NT, ND, NT, NH, NB * NH, 1.0f, 0, false);

    // x += y @ attnp_w + attnp_b
    gemm(y, 0, 0, NC,
         apT + (size_t)l * NC * NC, 0, 0, NC,
         attnp_b + l * NC, x, nullptr, x,
         0, 0, NC, BT, NC, NC, 1, 1, 1.0f, 0, false);

    layernorm_kernel<<<BT, 256, 0, stream>>>(x, ln2_w + l * NC, ln2_b + l * NC, h);

    // ff = gelu(h @ fc_w + fc_b)  -> bf16 [BT, FF]
    gemm(h, 0, 0, NC,
         fcT + (size_t)l * NFF * NC, 0, 0, NC,
         fc_b + l * NFF, nullptr, ff, nullptr,
         0, 0, NFF, BT, NFF, NC, 1, 1, 1.0f, 1, false);

    // x += ff @ fcp_w + fcp_b
    gemm(ff, 0, 0, NFF,
         fpT + (size_t)l * NC * NFF, 0, 0, NFF,
         fcp_b + l * NC, x, nullptr, x,
         0, 0, NC, BT, NC, NFF, 1, 1, 1.0f, 0, false);
  }

  // ---- final LN + LM head (logits straight into d_out, fp32) ----
  layernorm_kernel<<<BT, 256, 0, stream>>>(x, lnf_w, lnf_b, h);
  gemm(h, 0, 0, NC,
       lmT, 0, 0, NC,
       nullptr, out, nullptr, nullptr,
       0, 0, NV, BT, NV, NC, 1, 1, 1.0f, 0, true);

  // ---- cross entropy loss -> out[BT*NV] ----
  zero_accum_kernel<<<1, 2, 0, stream>>>(acc);
  ce_row_kernel<<<BT, 256, 0, stream>>>(out, targets, acc);
  finalize_loss_kernel<<<1, 1, 0, stream>>>(acc, out + (size_t)BT * NV);
}
